// VAD_Model_13932873908347
// MI455X (gfx1250) — matrix operand order repaired and checked by reading, not compiled, not run on hardware
//
#include <hip/hip_runtime.h>
#include <hip/hip_bf16.h>

#define T_STEPS 300
#define F_IN    39
#define H1S     35
#define H2S     30
#define BATCH   4096

#define NT1 10   // layer-1 N tiles (140 -> padded 160, split 5+5 across 2 waves)
#define NT2 8    // layer-2 N tiles (120 -> 128, split 4+4)
#define KT12 3   // both layers: K padded to 96 = 3 x 32
#define ZSTR 160 // z LDS row stride (floats)

typedef __attribute__((ext_vector_type(16))) __bf16 v16bf;
typedef __attribute__((ext_vector_type(8)))  float  v8f;

union Frag { v16bf v; uint4 q[2]; unsigned short us[16]; };

__device__ __forceinline__ unsigned short f2bf(float f) {
  unsigned u = __float_as_uint(f);
  unsigned r = (u + 0x7FFFu + ((u >> 16) & 1u)) >> 16;   // RNE
  return (unsigned short)r;
}

__device__ __forceinline__ float sigmoidf_(float x) { return 1.0f / (1.0f + __expf(-x)); }

// ---- A fragment (16x32 bf16) from LDS staging buffer a[16][96] (ushort bf16 bits).
// ISA 16-bit A layout: lanes 0-15 (row M=lane) hold K {0..7} in v0..3, {16..23} in v4..7;
// lanes 16-31 (row M=lane-16) hold K {8..15}, {24..31}. Both runs are 16B-aligned LDS b128s.
__device__ __forceinline__ v16bf load_a_frag(const unsigned short* a, int lane, int kt) {
  const int m    = lane & 15;
  const int base = m * 96 + kt * 32 + ((lane & 16) ? 8 : 0);
  Frag f;
  f.q[0] = *(const uint4*)(a + base);
  f.q[1] = *(const uint4*)(a + base + 16);
  return f.v;
}

// ---- B fragment (32x16 bf16) pre-packed in global: [nt][kt][lane][16 bf16] (32B/lane).
__device__ __forceinline__ v16bf load_b_frag(const unsigned short* wp, int lane, int kt, int nt) {
  const uint4* p = (const uint4*)(wp + (size_t)((nt * KT12 + kt) * 32 + lane) * 16);
  Frag f;
  f.q[0] = p[0];
  f.q[1] = p[1];
  return f.v;
}

// z[16][ZSTR] (f32, LDS) = A @ W + bias; this wave computes NTW consecutive n-tiles from
// ntBase. ntBase is wave-uniform DATA (not control flow) so both waves run an identical
// instruction stream -> EXEC stays all-ones at every WMMA. Per-lane bias values arrive
// pre-loaded in a register array (bv) so the T loop contains no global bias loads.
template<int NTW>
__device__ __forceinline__ void wave_gemm_bf16(const unsigned short* a,
                                               const unsigned short* __restrict__ wp,
                                               const float* bv,   // [NTW] in registers
                                               float* z, int lane, int ntBase) {
  const int col = lane & 15;
  const int r0  = (lane & 16) ? 8 : 0;
#pragma unroll
  for (int j = 0; j < NTW; ++j) {
    const int nt = ntBase + j;
    v8f acc = {0.f, 0.f, 0.f, 0.f, 0.f, 0.f, 0.f, 0.f};
#pragma unroll
    for (int kt = 0; kt < KT12; ++kt) {
      v16bf A = load_a_frag(a, lane, kt);
      v16bf B = load_b_frag(wp, lane, kt, nt);
      acc = __builtin_amdgcn_wmma_f32_16x16x32_bf16(false, A, false, B, (short)0, acc, false, false);
    }
    const int n = nt * 16 + col;
#pragma unroll
    for (int r = 0; r < 8; ++r) z[(r0 + r) * ZSTR + n] = acc[r] + bv[j];
  }
}

// ---- Pack fp32 row-major W (Kr x Nr) into WMMA B-fragment layout, bf16, zero-padded.
__global__ __launch_bounds__(64)
void pack_w_kernel(const float* __restrict__ W, unsigned short* __restrict__ out,
                   int Kr, int Nr, int KT, int NT) {
  int idx = blockIdx.x * 64 + threadIdx.x;
  int total = NT * KT * 32;
  if (idx >= total) return;
  int lane  = idx & 31;
  int ktnt  = idx >> 5;
  int kt    = ktnt % KT;
  int nt    = ktnt / KT;
  int n     = nt * 16 + (lane & 15);
  int kbase = kt * 32 + ((lane & 16) ? 8 : 0);
  unsigned short* o = out + (size_t)idx * 16;
#pragma unroll
  for (int h = 0; h < 16; ++h) {
    int k   = kbase + (h & 7) + ((h & 8) << 1);
    float v = (k < Kr && n < Nr) ? W[(size_t)k * Nr + n] : 0.0f;
    o[h] = f2bf(v);
  }
}

// ---- Fused bidirectional stacked LSTM. One block (2 waves) = one 16-row batch tile.
// The 2 waves split the N-tiles of each GEMM and the gate work; workgroup barriers
// order the shared LDS state. gridDim = (BATCH/16, 2 directions), blockDim = 64.
__global__ __launch_bounds__(64)
void lstm_bidir_kernel(const float* __restrict__ x,
                       const unsigned short* __restrict__ wfw1p, const float* __restrict__ bfw1,
                       const unsigned short* __restrict__ wfw2p, const float* __restrict__ bfw2,
                       const unsigned short* __restrict__ wbw1p, const float* __restrict__ bbw1,
                       const unsigned short* __restrict__ wbw2p, const float* __restrict__ bbw2,
                       _Float16* __restrict__ fwh, _Float16* __restrict__ bwh) {
  const int tid   = threadIdx.x;          // 0..63
  const int wave  = tid >> 5;             // 0 or 1 (wave-uniform)
  const int lane  = tid & 31;
  const int dir   = blockIdx.y;
  const int bbase = blockIdx.x * 16;

  const unsigned short* wp1 = dir ? wbw1p : wfw1p;
  const unsigned short* wp2 = dir ? wbw2p : wfw2p;
  const float* b1 = dir ? bbw1 : bfw1;
  const float* b2 = dir ? bbw2 : bfw2;
  _Float16* hout  = dir ? bwh  : fwh;

  __shared__ unsigned short a1[16 * 96];   // [x_t(39) | h1(35) | pad] bf16
  __shared__ unsigned short a2[16 * 96];   // [h1(35)  | h2(30) | pad] bf16
  __shared__ float          z [16 * ZSTR]; // gate pre-activations
  __shared__ float          c1[16 * H1S];
  __shared__ float          c2[16 * H2S];

  for (int i = tid; i < 16 * 96; i += 64) { a1[i] = 0; a2[i] = 0; }
  for (int i = tid; i < 16 * H1S; i += 64) c1[i] = 0.0f;
  for (int i = tid; i < 16 * H2S; i += 64) c2[i] = 0.0f;

  const int ntBase1 = wave * (NT1 / 2);    // 0 or 5
  const int ntBase2 = wave * (NT2 / 2);    // 0 or 4
  const int col     = lane & 15;

  // hoist per-lane bias values into registers (T loop stays free of global loads)
  float bv1[NT1 / 2], bv2[NT2 / 2];
#pragma unroll
  for (int j = 0; j < NT1 / 2; ++j) {
    int n = (ntBase1 + j) * 16 + col;
    bv1[j] = (n < 4 * H1S) ? b1[n] : 0.0f;
  }
#pragma unroll
  for (int j = 0; j < NT2 / 2; ++j) {
    int n = (ntBase2 + j) * 16 + col;
    bv2[j] = (n < 4 * H2S) ? b2[n] : 0.0f;
  }
  __syncthreads();

  for (int tt = 0; tt < T_STEPS; ++tt) {
    const int t = dir ? (T_STEPS - 1 - tt) : tt;

    // stage x_t tile into a1[:, 0:39]  (both waves split the work)
    for (int i = tid; i < 16 * F_IN; i += 64) {
      int m = i / F_IN, c = i % F_IN;
      a1[m * 96 + c] = f2bf(x[((size_t)(bbase + m) * T_STEPS + t) * F_IN + c]);
    }
    // prefetch next timestep's x tile into near caches (gfx1250 global_prefetch)
    {
      const int tn = dir ? (t - 1) : (t + 1);
      if (tn >= 0 && tn < T_STEPS) {
        const float* px = &x[((size_t)(bbase + (tid & 15)) * T_STEPS + tn) * F_IN];
        __builtin_prefetch(px, 0, 3);
        __builtin_prefetch((const char*)px + 128, 0, 3);
      }
    }
    __syncthreads();   // a1 x-region ready

    // layer 1: z = [x_t, h1] @ W1 + b1   (K=74->96, N=140->160, 5 n-tiles per wave)
    wave_gemm_bf16<NT1 / 2>(a1, wp1, bv1, z, lane, ntBase1);
    __syncthreads();   // z ready

    // layer 1 gates (split across both waves)
    for (int i = tid; i < 16 * H1S; i += 64) {
      int m = i / H1S, ci = i % H1S;
      const float* zr = z + m * ZSTR;
      float ig = sigmoidf_(zr[ci]);
      float jg = tanhf(zr[H1S + ci]);
      float fg = sigmoidf_(zr[2 * H1S + ci] + 1.0f);
      float og = sigmoidf_(zr[3 * H1S + ci]);
      float c  = fg * c1[i] + ig * jg;
      c1[i] = c;
      float h = og * tanhf(c);
      unsigned short hb = f2bf(h);
      a1[m * 96 + F_IN + ci] = hb;   // recurrent input for layer 1
      a2[m * 96 + ci]        = hb;   // feed layer 2
    }
    __syncthreads();   // a1 h-region / a2 h1-region / c1 ready

    // layer 2: z = [h1, h2] @ W2 + b2   (K=65->96, N=120->128, 4 n-tiles per wave)
    wave_gemm_bf16<NT2 / 2>(a2, wp2, bv2, z, lane, ntBase2);
    __syncthreads();   // z ready

    // layer 2 gates + emit h2 (split across both waves)
    for (int i = tid; i < 16 * H2S; i += 64) {
      int m = i / H2S, ci = i % H2S;
      const float* zr = z + m * ZSTR;
      float ig = sigmoidf_(zr[ci]);
      float jg = tanhf(zr[H2S + ci]);
      float fg = sigmoidf_(zr[2 * H2S + ci] + 1.0f);
      float og = sigmoidf_(zr[3 * H2S + ci]);
      float c  = fg * c2[i] + ig * jg;
      c2[i] = c;
      float h = og * tanhf(c);
      a2[m * 96 + H1S + ci] = f2bf(h);
      hout[((size_t)(bbase + m) * T_STEPS + t) * 32 + ci] = (_Float16)h;
    }
    // loop-top barrier (after stage) covers gates2 -> next-iteration hazards
  }
}

// ---- MLP head: one thread per (b,t). Weights cached in LDS, fully unrolled.
__global__ __launch_bounds__(256)
void fc_kernel(const _Float16* __restrict__ fwh, const _Float16* __restrict__ bwh,
               const float* __restrict__ w1, const float* __restrict__ bb1,
               const float* __restrict__ w2, const float* __restrict__ bb2,
               const float* __restrict__ w3, const float* __restrict__ bb3,
               float* __restrict__ out) {
  __shared__ float sw1[60 * 50];
  __shared__ float sw2[50 * 40];
  __shared__ float sw3[40 * 2];
  __shared__ float sb1[50], sb2[40], sb3[2];
  for (int i = threadIdx.x; i < 60 * 50; i += 256) sw1[i] = w1[i];
  for (int i = threadIdx.x; i < 50 * 40; i += 256) sw2[i] = w2[i];
  for (int i = threadIdx.x; i < 80;      i += 256) sw3[i] = w3[i];
  for (int i = threadIdx.x; i < 50;      i += 256) sb1[i] = bb1[i];
  for (int i = threadIdx.x; i < 40;      i += 256) sb2[i] = bb2[i];
  if (threadIdx.x < 2) sb3[threadIdx.x] = bb3[threadIdx.x];
  __syncthreads();

  size_t idx = (size_t)blockIdx.x * 256 + threadIdx.x;   // = b*T + t
  if (idx >= (size_t)BATCH * T_STEPS) return;
  const _Float16* fr = fwh + idx * 32;
  const _Float16* br = bwh + idx * 32;

  float in[60];
#pragma unroll
  for (int k = 0; k < 30; ++k) { in[k] = (float)fr[k]; in[30 + k] = (float)br[k]; }

  float h1[50];
#pragma unroll
  for (int j = 0; j < 50; ++j) {
    float s = sb1[j];
#pragma unroll
    for (int k = 0; k < 60; ++k) s = fmaf(in[k], sw1[k * 50 + j], s);
    h1[j] = fmaxf(s, 0.0f);
  }
  float h2[40];
#pragma unroll
  for (int j = 0; j < 40; ++j) {
    float s = sb2[j];
#pragma unroll
    for (int k = 0; k < 50; ++k) s = fmaf(h1[k], sw2[k * 40 + j], s);
    h2[j] = fmaxf(s, 0.0f);
  }
  float o0 = sb3[0], o1 = sb3[1];
#pragma unroll
  for (int k = 0; k < 40; ++k) {
    o0 = fmaf(h2[k], sw3[k * 2 + 0], o0);
    o1 = fmaf(h2[k], sw3[k * 2 + 1], o1);
  }
  out[idx * 2 + 0] = o0;
  out[idx * 2 + 1] = o1;
}

extern "C" void kernel_launch(void* const* d_in, const int* in_sizes, int n_in,
                              void* d_out, int out_size, void* d_ws, size_t ws_size,
                              hipStream_t stream) {
  (void)in_sizes; (void)n_in; (void)out_size; (void)ws_size;
  const float* x    = (const float*)d_in[0];
  const float* wfw1 = (const float*)d_in[1];
  const float* bfw1 = (const float*)d_in[2];
  const float* wfw2 = (const float*)d_in[3];
  const float* bfw2 = (const float*)d_in[4];
  const float* wbw1 = (const float*)d_in[5];
  const float* bbw1 = (const float*)d_in[6];
  const float* wbw2 = (const float*)d_in[7];
  const float* bbw2 = (const float*)d_in[8];
  const float* wfc1 = (const float*)d_in[9];
  const float* bfc1 = (const float*)d_in[10];
  const float* wfc2 = (const float*)d_in[11];
  const float* bfc2 = (const float*)d_in[12];
  const float* wfc3 = (const float*)d_in[13];
  const float* bfc3 = (const float*)d_in[14];

  char* ws = (char*)d_ws;
  size_t off = 0;
  auto carve = [&](size_t bytes) -> void* {
    void* p = ws + off;
    off = (off + bytes + 255) & ~(size_t)255;
    return p;
  };
  unsigned short* p_fw1 = (unsigned short*)carve((size_t)NT1 * KT12 * 32 * 16 * 2);
  unsigned short* p_fw2 = (unsigned short*)carve((size_t)NT2 * KT12 * 32 * 16 * 2);
  unsigned short* p_bw1 = (unsigned short*)carve((size_t)NT1 * KT12 * 32 * 16 * 2);
  unsigned short* p_bw2 = (unsigned short*)carve((size_t)NT2 * KT12 * 32 * 16 * 2);
  _Float16* fwh = (_Float16*)carve((size_t)BATCH * T_STEPS * 32 * 2);
  _Float16* bwh = (_Float16*)carve((size_t)BATCH * T_STEPS * 32 * 2);

  // pack weights into WMMA B-fragment layout (bf16, zero-padded)
  pack_w_kernel<<<(NT1 * KT12 * 32 + 63) / 64, 64, 0, stream>>>(wfw1, p_fw1, 74, 140, KT12, NT1);
  pack_w_kernel<<<(NT2 * KT12 * 32 + 63) / 64, 64, 0, stream>>>(wfw2, p_fw2, 65, 120, KT12, NT2);
  pack_w_kernel<<<(NT1 * KT12 * 32 + 63) / 64, 64, 0, stream>>>(wbw1, p_bw1, 74, 140, KT12, NT1);
  pack_w_kernel<<<(NT2 * KT12 * 32 + 63) / 64, 64, 0, stream>>>(wbw2, p_bw2, 65, 120, KT12, NT2);

  // fused bidirectional stacked LSTM (both directions in one grid)
  lstm_bidir_kernel<<<dim3(BATCH / 16, 2), 64, 0, stream>>>(
      x, p_fw1, bfw1, p_fw2, bfw2, p_bw1, bbw1, p_bw2, bbw2, fwh, bwh);

  // MLP head
  size_t rows = (size_t)BATCH * T_STEPS;
  fc_kernel<<<(unsigned)((rows + 255) / 256), 256, 0, stream>>>(
      fwh, bwh, wfc1, bfc1, wfc2, bfc2, wfc3, bfc3, (float*)d_out);
}
